// ConceptualSparseAttention_77472620085360
// MI455X (gfx1250) — compile-verified, matrix operand order checked by reading, not run on hardware
//
#include <hip/hip_runtime.h>
#include <hip/hip_bf16.h>

// Problem constants (from reference)
#define B_   2
#define S_   2048
#define D_   512
#define H_   8
#define HD_  64
#define KTOP_ 307
#define HW_  16        // half window
#define RC_  16        // random links per query

typedef __attribute__((ext_vector_type(16))) __bf16        v16bf;
typedef __attribute__((ext_vector_type(8)))  float          v8f;
typedef __attribute__((ext_vector_type(8)))  unsigned short u16x8;

// ---------- helpers ----------

static __device__ __forceinline__ unsigned short f2bf(float f) {
  unsigned u = __float_as_uint(f);
  unsigned r = u + 0x7FFFu + ((u >> 16) & 1u);   // RNE
  return (unsigned short)(r >> 16);
}

static __device__ __forceinline__ v16bf mk16(u16x8 a, u16x8 b) {
  union { unsigned short s[16]; v16bf v; } t;
#pragma unroll
  for (int i = 0; i < 8; ++i) { t.s[i] = a[i]; t.s[8 + i] = b[i]; }
  return t.v;
}

static __device__ __forceinline__ u16x8 ld8(const unsigned short* p) {
  return *(const u16x8*)p;
}

static __device__ __forceinline__ v8f zero8() {
  v8f c;
#pragma unroll
  for (int i = 0; i < 8; ++i) c[i] = 0.f;
  return c;
}

// 16x64 output macro-tile GEMM over K=512 (bf16 WMMA, fp32 acc).
// Software-pipelined: fragments for step k+32 are fetched before the
// step-k WMMAs issue, so matrix math overlaps the VMEM latency and the
// compiler can use partial s_wait_loadcnt instead of a full drain.
static __device__ __forceinline__ void gemm_tile4(const unsigned short* __restrict__ arow,
                                                  const unsigned short* __restrict__ brow,
                                                  int grp0,
                                                  v8f& c0, v8f& c1, v8f& c2, v8f& c3) {
  const unsigned short* b0 = brow;
  const unsigned short* b1 = brow + 16 * D_;
  const unsigned short* b2 = brow + 32 * D_;
  const unsigned short* b3 = brow + 48 * D_;

  v16bf a  = mk16(ld8(arow + grp0), ld8(arow + grp0 + 16));
  v16bf f0 = mk16(ld8(b0 + grp0), ld8(b0 + grp0 + 16));
  v16bf f1 = mk16(ld8(b1 + grp0), ld8(b1 + grp0 + 16));
  v16bf f2 = mk16(ld8(b2 + grp0), ld8(b2 + grp0 + 16));
  v16bf f3 = mk16(ld8(b3 + grp0), ld8(b3 + grp0 + 16));

#pragma unroll
  for (int k = 0; k < D_; k += 32) {
    v16bf an = a, g0 = f0, g1 = f1, g2 = f2, g3 = f3;
    if (k + 32 < D_) {
      const int kn = k + 32;
      __builtin_prefetch((const void*)(arow + kn + 64), 0, 3);
      __builtin_prefetch((const void*)(b0 + kn + 64), 0, 3);
      __builtin_prefetch((const void*)(b1 + kn + 64), 0, 3);
      __builtin_prefetch((const void*)(b2 + kn + 64), 0, 3);
      __builtin_prefetch((const void*)(b3 + kn + 64), 0, 3);
      an = mk16(ld8(arow + kn + grp0), ld8(arow + kn + grp0 + 16));
      g0 = mk16(ld8(b0 + kn + grp0), ld8(b0 + kn + grp0 + 16));
      g1 = mk16(ld8(b1 + kn + grp0), ld8(b1 + kn + grp0 + 16));
      g2 = mk16(ld8(b2 + kn + grp0), ld8(b2 + kn + grp0 + 16));
      g3 = mk16(ld8(b3 + kn + grp0), ld8(b3 + kn + grp0 + 16));
    }
    c0 = __builtin_amdgcn_wmma_f32_16x16x32_bf16(false, a, false, f0, (short)0, c0, false, false);
    c1 = __builtin_amdgcn_wmma_f32_16x16x32_bf16(false, a, false, f1, (short)0, c1, false, false);
    c2 = __builtin_amdgcn_wmma_f32_16x16x32_bf16(false, a, false, f2, (short)0, c2, false, false);
    c3 = __builtin_amdgcn_wmma_f32_16x16x32_bf16(false, a, false, f3, (short)0, c3, false, false);
    a = an; f0 = g0; f1 = g1; f2 = g2; f3 = g3;
  }
}

// ---------- conversion kernels ----------

__global__ void f32_to_bf16_kernel(const float* __restrict__ src,
                                   unsigned short* __restrict__ dst, int n) {
  for (int i = blockIdx.x * blockDim.x + threadIdx.x; i < n; i += gridDim.x * blockDim.x)
    dst[i] = f2bf(src[i]);
}

// Wt[n][k] = W[k][n], 512x512, bf16 output
__global__ void transpose_w_kernel(const float* __restrict__ W,
                                   unsigned short* __restrict__ Wt) {
  int e = blockIdx.x * blockDim.x + threadIdx.x;  // 262144 threads
  int n = e & 511, k = e >> 9;
  Wt[n * 512 + k] = f2bf(W[k * 512 + n]);
}

// ---------- importance scorer: sigmoid(relu(x@Ws1+bs1)@Ws2+bs2) ----------

__global__ void __launch_bounds__(256)
scorer_kernel(const float* __restrict__ x, const float* __restrict__ Ws1,
              const float* __restrict__ bs1, const float* __restrict__ Ws2,
              const float* __restrict__ bs2, float* __restrict__ imp) {
  __shared__ float xs[D_];
  __shared__ float red[256];
  const int tok = blockIdx.x;
  const int tid = threadIdx.x;
  const float* xr = x + tok * D_;
  for (int i = tid; i < D_; i += 256) xs[i] = xr[i];
  __syncthreads();
  float acc = 0.f;
  for (int k = 0; k < D_; ++k) acc += xs[k] * Ws1[k * 256 + tid];
  float hcol = fmaxf(acc + bs1[tid], 0.f);
  red[tid] = hcol * Ws2[tid];
  __syncthreads();
  for (int s = 128; s > 0; s >>= 1) {
    if (tid < s) red[tid] += red[tid + s];
    __syncthreads();
  }
  if (tid == 0) imp[tok] = 1.f / (1.f + __expf(-(red[0] + bs2[0])));
}

// ---------- top-K threshold via binary search on monotone uint image ----------

__global__ void __launch_bounds__(256)
topk_threshold_kernel(const float* __restrict__ imp, float* __restrict__ thr) {
  __shared__ unsigned su[S_];
  __shared__ int cnt;
  const int b = blockIdx.x, tid = threadIdx.x;
  for (int i = tid; i < S_; i += 256) su[i] = __float_as_uint(imp[b * S_ + i]);
  __syncthreads();
  unsigned lo = 0u, hi = 0x3F800000u;  // sigmoid in (0,1)
  for (int iter = 0; iter < 32; ++iter) {
    unsigned mid = lo + ((hi - lo) >> 1);
    if (mid == lo) break;              // uniform across block
    if (tid == 0) cnt = 0;
    __syncthreads();
    int c = 0;
    for (int i = tid; i < S_; i += 256) c += (su[i] >= mid) ? 1 : 0;
    atomicAdd(&cnt, c);
    __syncthreads();
    int total = cnt;
    __syncthreads();
    if (total >= KTOP_) lo = mid; else hi = mid;
  }
  if (tid == 0) thr[b] = __uint_as_float(lo);
}

// ---------- mask bitwords: maskw[b][i][w] covers j in [32w, 32w+32) ----------

__global__ void __launch_bounds__(256)
mask_kernel(const float* __restrict__ imp, const float* __restrict__ thr,
            const int* __restrict__ rand_idx, unsigned* __restrict__ maskw) {
  const int wave = (blockIdx.x * 256 + threadIdx.x) >> 5;
  const int lane = threadIdx.x & 31;
  const int b = wave >> 11;
  const int i = wave & 2047;
  const int dense = imp[b * S_ + i] >= thr[b];
  int rj[RC_];
  const int* rp = rand_idx + (b * S_ + i) * RC_;
#pragma unroll
  for (int c = 0; c < RC_; ++c) rj[c] = rp[c];
  unsigned* mrow = maskw + (b * S_ + i) * 64;
#pragma unroll
  for (int half = 0; half < 2; ++half) {
    const int w = lane + half * 32;
    const int jbase = w * 32;
    unsigned causal;
    if (i >= jbase + 31)      causal = 0xFFFFFFFFu;
    else if (i < jbase)       causal = 0u;
    else                      causal = 0xFFFFFFFFu >> (31 - (i - jbase));
    unsigned mm;
    if (dense) {
      mm = 0xFFFFFFFFu;
    } else {
      mm = 0u;
      int lw = i - HW_, hw = i + HW_;
      int s = lw > jbase ? lw : jbase;
      int e2 = hw < jbase + 31 ? hw : jbase + 31;
      if (s <= e2) {
        int len = e2 - s + 1;
        unsigned bits = (len >= 32) ? 0xFFFFFFFFu : ((1u << len) - 1u);
        mm |= bits << (s - jbase);
      }
#pragma unroll
      for (int c = 0; c < RC_; ++c) {
        int j = rj[c];
        if ((j >> 5) == w) mm |= 1u << (j & 31);
      }
    }
    mrow[w] = mm & causal;
  }
}

// ---------- per (i-tile16, j-tile16) occupancy ----------

__global__ void jany_kernel(const unsigned* __restrict__ maskw,
                            unsigned char* __restrict__ jany) {
  const int idx = blockIdx.x * 256 + threadIdx.x;   // B*128*128
  const int jt = idx & 127;
  const int it = (idx >> 7) & 127;
  const int b = idx >> 14;
  unsigned acc = 0;
#pragma unroll
  for (int r = 0; r < 16; ++r)
    acc |= maskw[(b * S_ + it * 16 + r) * 64 + (jt >> 1)];
  unsigned nib = (acc >> ((jt & 1) * 16)) & 0xFFFFu;
  jany[idx] = nib ? 1 : 0;
}

// ---------- fused QKV projection (bf16 WMMA), K head-major, V transposed ----------
// One wave computes a 16x64 macro-tile; the 64-wide n-group == one head.

__global__ void __launch_bounds__(256)
gemm_qkv_kernel(const unsigned short* __restrict__ xb,
                const unsigned short* __restrict__ Wqt,
                const unsigned short* __restrict__ Wkt,
                const unsigned short* __restrict__ Wvt,
                const float* __restrict__ bq, const float* __restrict__ bk,
                const float* __restrict__ bv,
                unsigned short* __restrict__ Qb, unsigned short* __restrict__ Kb,
                unsigned short* __restrict__ Vtb) {
  const int wave = (blockIdx.x * 256 + threadIdx.x) >> 5;   // 3 * 256 * 8 = 6144 waves
  const int lane = threadIdx.x & 31;
  const int which = wave >> 11;       // 2048 macro-tiles per matrix
  const int t = wave & 2047;
  const int mt = t >> 3;              // 256 row tiles
  const int ng = t & 7;               // 8 n-groups of 64 == head index
  const int m0 = mt * 16;
  const int ln = lane & 15, hi = lane >> 4, grp0 = hi * 8;
  const unsigned short* Wt = (which == 0) ? Wqt : (which == 1 ? Wkt : Wvt);
  v8f c0 = zero8(), c1 = zero8(), c2 = zero8(), c3 = zero8();
  gemm_tile4(xb + (m0 + ln) * D_, Wt + (ng * 64 + ln) * D_, grp0, c0, c1, c2, c3);

  const float* bias = (which == 0) ? bq : (which == 1 ? bk : bv);
  const int b = m0 >> 11;
  const int i0 = m0 & 2047;
  const int h = ng;
  float bb[4];
#pragma unroll
  for (int tt = 0; tt < 4; ++tt) bb[tt] = bias[ng * 64 + tt * 16 + ln];

  if (which < 2) {
    unsigned short* dst = ((which == 0) ? Qb : Kb) + ((b * H_ + h) * S_) * HD_;
#pragma unroll
    for (int e = 0; e < 8; ++e) {
      const int i = i0 + e + hi * 8;
      dst[i * HD_ + 0 + ln]  = f2bf(c0[e] + bb[0]);
      dst[i * HD_ + 16 + ln] = f2bf(c1[e] + bb[1]);
      dst[i * HD_ + 32 + ln] = f2bf(c2[e] + bb[2]);
      dst[i * HD_ + 48 + ln] = f2bf(c3[e] + bb[3]);
    }
  } else {
    unsigned short* dstb = Vtb + ((b * H_ + h) * HD_) * S_;
#pragma unroll
    for (int e = 0; e < 8; ++e) {
      const int i = i0 + e + hi * 8;
      dstb[(0 + ln) * S_ + i]  = f2bf(c0[e] + bb[0]);
      dstb[(16 + ln) * S_ + i] = f2bf(c1[e] + bb[1]);
      dstb[(32 + ln) * S_ + i] = f2bf(c2[e] + bb[2]);
      dstb[(48 + ln) * S_ + i] = f2bf(c3[e] + bb[3]);
    }
  }
}

// ---------- sparse flash attention: one wave per (b, h, 16-row i-tile) ----------

__global__ void __launch_bounds__(256)
attn_kernel(const unsigned short* __restrict__ Qb,
            const unsigned short* __restrict__ Kb,
            const unsigned short* __restrict__ Vtb,
            const unsigned* __restrict__ maskw,
            const unsigned char* __restrict__ jany,
            unsigned short* __restrict__ attnO) {
  __shared__ __align__(16) unsigned short Plds[8 * 16 * 32];  // 8 waves x (16x32 bf16)
  const int lane = threadIdx.x & 31;
  const int wv = threadIdx.x >> 5;
  const int wg = blockIdx.x * 8 + wv;     // 2048 waves total
  const int it = wg & 127;
  const int h = (wg >> 7) & 7;
  const int b = wg >> 10;
  const int i0 = it * 16;
  const int ln = lane & 15;
  const int hi = lane >> 4;
  const int grp0 = hi * 8;
  unsigned short* Pw = &Plds[wv * 512];

  const unsigned short* Qh = Qb + ((b * H_ + h) * S_) * HD_;
  const unsigned short* Kh = Kb + ((b * H_ + h) * S_) * HD_;
  const unsigned short* Vh = Vtb + ((b * H_ + h) * HD_) * S_;

  // Q A-fragments (rows i0..i0+15, d in [0,32) and [32,64)), loaded once
  const unsigned short* qrow = Qh + (i0 + ln) * HD_;
  const v16bf aq0 = mk16(ld8(qrow + grp0), ld8(qrow + grp0 + 16));
  const v16bf aq1 = mk16(ld8(qrow + 32 + grp0), ld8(qrow + 32 + grp0 + 16));

  v8f o0 = zero8(), o1 = zero8(), o2 = zero8(), o3 = zero8();
  float m[8], l[8];
#pragma unroll
  for (int e = 0; e < 8; ++e) { m[e] = -__builtin_inff(); l[e] = 0.f; }

  const int nblk = (it >> 1) + 1;                     // 32-wide j blocks, causal
  const unsigned char* janyRow = jany + ((b * 128 + it) << 7);

  for (int jb = 0; jb < nblk; ++jb) {
    const int jt0 = jb * 2, jt1 = jb * 2 + 1;
    const int a0 = janyRow[jt0];
    const int a1 = janyRow[jt1];
    if (!(a0 | a1)) continue;                         // structural sparsity skip

    unsigned mw[8];
#pragma unroll
    for (int e = 0; e < 8; ++e) {
      int i = i0 + e + hi * 8;
      mw[e] = maskw[(b * S_ + i) * 64 + jb];          // word == 32 j's of this block
    }

    float s0[8], s1[8];
    if (a0) {
      const unsigned short* krow = Kh + (jt0 * 16 + ln) * HD_;
      v16bf bk0 = mk16(ld8(krow + grp0), ld8(krow + grp0 + 16));
      v16bf bk1 = mk16(ld8(krow + 32 + grp0), ld8(krow + 32 + grp0 + 16));
      v8f c = zero8();
      c = __builtin_amdgcn_wmma_f32_16x16x32_bf16(false, aq0, false, bk0, (short)0, c, false, false);
      c = __builtin_amdgcn_wmma_f32_16x16x32_bf16(false, aq1, false, bk1, (short)0, c, false, false);
#pragma unroll
      for (int e = 0; e < 8; ++e)
        s0[e] = ((mw[e] >> ln) & 1u) ? c[e] * 0.125f : -__builtin_inff();
    } else {
#pragma unroll
      for (int e = 0; e < 8; ++e) s0[e] = -__builtin_inff();
    }
    if (a1) {
      const unsigned short* krow = Kh + (jt1 * 16 + ln) * HD_;
      v16bf bk0 = mk16(ld8(krow + grp0), ld8(krow + grp0 + 16));
      v16bf bk1 = mk16(ld8(krow + 32 + grp0), ld8(krow + 32 + grp0 + 16));
      v8f c = zero8();
      c = __builtin_amdgcn_wmma_f32_16x16x32_bf16(false, aq0, false, bk0, (short)0, c, false, false);
      c = __builtin_amdgcn_wmma_f32_16x16x32_bf16(false, aq1, false, bk1, (short)0, c, false, false);
#pragma unroll
      for (int e = 0; e < 8; ++e)
        s1[e] = ((mw[e] >> (ln + 16)) & 1u) ? c[e] * 0.125f : -__builtin_inff();
    } else {
#pragma unroll
      for (int e = 0; e < 8; ++e) s1[e] = -__builtin_inff();
    }

    // online softmax across the 32 j's of this block (row = 16 lanes of one half)
    float sc[8];
#pragma unroll
    for (int e = 0; e < 8; ++e) {
      float mx = fmaxf(s0[e], s1[e]);
#pragma unroll
      for (int off = 1; off < 16; off <<= 1)
        mx = fmaxf(mx, __shfl_xor(mx, off, 32));
      float mn = fmaxf(m[e], mx);
      float scl, p0, p1;
      if (mn == -__builtin_inff()) {
        scl = 1.f; p0 = 0.f; p1 = 0.f;
      } else {
        scl = __expf(m[e] - mn);                     // m may be -inf -> 0
        p0 = (s0[e] == -__builtin_inff()) ? 0.f : __expf(s0[e] - mn);
        p1 = (s1[e] == -__builtin_inff()) ? 0.f : __expf(s1[e] - mn);
      }
      float rs = p0 + p1;
#pragma unroll
      for (int off = 1; off < 16; off <<= 1)
        rs += __shfl_xor(rs, off, 32);
      l[e] = l[e] * scl + rs;
      m[e] = mn;
      sc[e] = scl;
      const int row = e + hi * 8;                    // stage P tile to LDS as bf16
      Pw[row * 32 + ln]      = f2bf(p0);
      Pw[row * 32 + ln + 16] = f2bf(p1);
    }
#pragma unroll
    for (int e = 0; e < 8; ++e) {
      o0[e] *= sc[e]; o1[e] *= sc[e]; o2[e] *= sc[e]; o3[e] *= sc[e];
    }

    __builtin_amdgcn_wave_barrier();
    asm volatile("s_wait_dscnt 0" ::: "memory");     // P stores visible for A-frag reads

    const unsigned short* prow = Pw + ln * 32;
    v16bf pA = mk16(ld8(prow + grp0), ld8(prow + grp0 + 16));

    const int jc = jb * 32;
    {
      const unsigned short* vr = Vh + (0 + ln) * S_ + jc;
      v16bf bvv = mk16(ld8(vr + grp0), ld8(vr + grp0 + 16));
      o0 = __builtin_amdgcn_wmma_f32_16x16x32_bf16(false, pA, false, bvv, (short)0, o0, false, false);
    }
    {
      const unsigned short* vr = Vh + (16 + ln) * S_ + jc;
      v16bf bvv = mk16(ld8(vr + grp0), ld8(vr + grp0 + 16));
      o1 = __builtin_amdgcn_wmma_f32_16x16x32_bf16(false, pA, false, bvv, (short)0, o1, false, false);
    }
    {
      const unsigned short* vr = Vh + (32 + ln) * S_ + jc;
      v16bf bvv = mk16(ld8(vr + grp0), ld8(vr + grp0 + 16));
      o2 = __builtin_amdgcn_wmma_f32_16x16x32_bf16(false, pA, false, bvv, (short)0, o2, false, false);
    }
    {
      const unsigned short* vr = Vh + (48 + ln) * S_ + jc;
      v16bf bvv = mk16(ld8(vr + grp0), ld8(vr + grp0 + 16));
      o3 = __builtin_amdgcn_wmma_f32_16x16x32_bf16(false, pA, false, bvv, (short)0, o3, false, false);
    }
  }

  // normalize and write bf16 output [b][i][h*64+d]
#pragma unroll
  for (int e = 0; e < 8; ++e) {
    const float inv = 1.f / l[e];                    // diagonal always allowed -> l > 0
    const int i = i0 + e + hi * 8;
    unsigned short* orow = attnO + (b * S_ + i) * D_ + h * HD_;
    orow[0 + ln]  = f2bf(o0[e] * inv);
    orow[16 + ln] = f2bf(o1[e] * inv);
    orow[32 + ln] = f2bf(o2[e] * inv);
    orow[48 + ln] = f2bf(o3[e] * inv);
  }
}

// ---------- final projection: out = attnO @ Wo + bo (fp32 out), 16x64 per wave ----------

__global__ void __launch_bounds__(256)
gemm_out_kernel(const unsigned short* __restrict__ Ab,
                const unsigned short* __restrict__ Wot,
                const float* __restrict__ bo, float* __restrict__ out) {
  const int wave = (blockIdx.x * 256 + threadIdx.x) >> 5;  // 256 * 8 = 2048 waves
  const int lane = threadIdx.x & 31;
  const int mt = wave >> 3, ng = wave & 7;
  const int m0 = mt * 16;
  const int ln = lane & 15, hi = lane >> 4, grp0 = hi * 8;
  v8f c0 = zero8(), c1 = zero8(), c2 = zero8(), c3 = zero8();
  gemm_tile4(Ab + (m0 + ln) * D_, Wot + (ng * 64 + ln) * D_, grp0, c0, c1, c2, c3);
  float bb[4];
#pragma unroll
  for (int tt = 0; tt < 4; ++tt) bb[tt] = bo[ng * 64 + tt * 16 + ln];
#pragma unroll
  for (int e = 0; e < 8; ++e) {
    const int mrow = m0 + e + hi * 8;
    float* orow = out + mrow * D_ + ng * 64;
    orow[0 + ln]  = c0[e] + bb[0];
    orow[16 + ln] = c1[e] + bb[1];
    orow[32 + ln] = c2[e] + bb[2];
    orow[48 + ln] = c3[e] + bb[3];
  }
}

// ---------- host launcher ----------

extern "C" void kernel_launch(void* const* d_in, const int* in_sizes, int n_in,
                              void* d_out, int out_size, void* d_ws, size_t ws_size,
                              hipStream_t stream) {
  const float* x    = (const float*)d_in[0];
  const float* Wq   = (const float*)d_in[1];
  const float* bq   = (const float*)d_in[2];
  const float* Wk   = (const float*)d_in[3];
  const float* bk   = (const float*)d_in[4];
  const float* Wv   = (const float*)d_in[5];
  const float* bv   = (const float*)d_in[6];
  const float* Wo   = (const float*)d_in[7];
  const float* bo   = (const float*)d_in[8];
  const float* Ws1  = (const float*)d_in[9];
  const float* bs1  = (const float*)d_in[10];
  const float* Ws2  = (const float*)d_in[11];
  const float* bs2  = (const float*)d_in[12];
  const int* rand_idx = (const int*)d_in[13];
  float* out = (float*)d_out;

  // workspace carving (256B aligned)
  char* ws = (char*)d_ws;
  size_t off = 0;
  auto carve = [&](size_t bytes) -> char* {
    char* p = ws + off;
    off = (off + bytes + 255) & ~(size_t)255;
    return p;
  };
  unsigned short* xb   = (unsigned short*)carve((size_t)B_ * S_ * D_ * 2);     // 4 MB
  unsigned short* Wqt  = (unsigned short*)carve((size_t)D_ * D_ * 2);
  unsigned short* Wkt  = (unsigned short*)carve((size_t)D_ * D_ * 2);
  unsigned short* Wvt  = (unsigned short*)carve((size_t)D_ * D_ * 2);
  unsigned short* Wot  = (unsigned short*)carve((size_t)D_ * D_ * 2);
  unsigned short* Qb   = (unsigned short*)carve((size_t)B_ * H_ * S_ * HD_ * 2);
  unsigned short* Kb   = (unsigned short*)carve((size_t)B_ * H_ * S_ * HD_ * 2);
  unsigned short* Vtb  = (unsigned short*)carve((size_t)B_ * H_ * HD_ * S_ * 2);
  unsigned short* aO   = (unsigned short*)carve((size_t)B_ * S_ * D_ * 2);
  float*          imp  = (float*)carve((size_t)B_ * S_ * 4);
  float*          thr  = (float*)carve(256);
  unsigned*       mskw = (unsigned*)carve((size_t)B_ * S_ * 64 * 4);           // 1 MB
  unsigned char*  jany = (unsigned char*)carve((size_t)B_ * 128 * 128);

  // 1. conversions
  f32_to_bf16_kernel<<<1024, 256, 0, stream>>>(x, xb, B_ * S_ * D_);
  transpose_w_kernel<<<1024, 256, 0, stream>>>(Wq, Wqt);
  transpose_w_kernel<<<1024, 256, 0, stream>>>(Wk, Wkt);
  transpose_w_kernel<<<1024, 256, 0, stream>>>(Wv, Wvt);
  transpose_w_kernel<<<1024, 256, 0, stream>>>(Wo, Wot);

  // 2. importance scores + top-K threshold
  scorer_kernel<<<B_ * S_, 256, 0, stream>>>(x, Ws1, bs1, Ws2, bs2, imp);
  topk_threshold_kernel<<<B_, 256, 0, stream>>>(imp, thr);

  // 3. mask bitwords + j-tile occupancy
  mask_kernel<<<(B_ * S_) / 8, 256, 0, stream>>>(imp, thr, rand_idx, mskw);
  jany_kernel<<<(B_ * 128 * 128) / 256, 256, 0, stream>>>(mskw, jany);

  // 4. QKV projections (3 * 2048 macro-tile waves, 8 waves/block)
  gemm_qkv_kernel<<<768, 256, 0, stream>>>(xb, Wqt, Wkt, Wvt, bq, bk, bv, Qb, Kb, Vtb);

  // 5. sparse flash attention (2048 waves)
  attn_kernel<<<256, 256, 0, stream>>>(Qb, Kb, Vtb, mskw, jany, aO);

  // 6. output projection (2048 macro-tile waves)
  gemm_out_kernel<<<256, 256, 0, stream>>>(aO, Wot, bo, out);

  (void)in_sizes; (void)n_in; (void)out_size; (void)ws_size;
}
